// OrthogonalRNN_77360950935624
// MI455X (gfx1250) — matrix operand order repaired and checked by reading, not compiled, not run on hardware
//
#include <hip/hip_runtime.h>

typedef __attribute__((ext_vector_type(16))) _Float16 v16h;
typedef __attribute__((ext_vector_type(8)))  _Float16 v8h;
typedef __attribute__((ext_vector_type(8)))  float    v8f;

#define RNN_B   1024
#define RNN_T   784
#define RNN_H   512
#define RNN_C   10
#define BQ      16          // batch rows per block
#define HSTR    520         // LDS h row stride in halves (bank stagger)
#define KHALF   256         // K-range of W_rec resident in LDS
#define WSTR    264         // LDS W row stride in halves (bank stagger)
#define NTHREADS 512        // 16 waves; each wave owns 2 column tiles

union HV { v16h v16; v8h v8[2]; };

// ---- prep: W_rec f32 -> f16 (row-major; rows of W_rec are WMMA-B K-vectors)
__global__ void wrec_to_f16(const float* __restrict__ wrec, _Float16* __restrict__ wh) {
    int idx = blockIdx.x * blockDim.x + threadIdx.x;
    if (idx < RNN_H * RNN_H) wh[idx] = (_Float16)wrec[idx];
}

// ---- prep: permute inputs into time-major xp[t][b]
__global__ void permute_x(const float* __restrict__ inp, const int* __restrict__ perm,
                          float* __restrict__ xp) {
    int idx = blockIdx.x * blockDim.x + threadIdx.x;
    if (idx < RNN_T * RNN_B) {
        int t = idx >> 10;
        int b = idx & (RNN_B - 1);
        xp[idx] = inp[b * RNN_T + perm[t]];
    }
}

// ---- main: each block owns 16 batch rows for the entire 784-step recurrence.
// LDS: W_rec K-half (270 KB) + h double buffer (33 KB) + w_in + x  ~= 306 KB/WGP.
__global__ __launch_bounds__(NTHREADS) void rnn_kernel(
        const float* __restrict__ xp,       // [T][1024]
        const _Float16* __restrict__ wh,    // [512][512] f16 W_rec (L2-resident)
        const float* __restrict__ win,      // [512]
        const float* __restrict__ wout,     // [10][512]
        const float* __restrict__ bout,     // [10]
        float* __restrict__ out)            // [1024][10]
{
    extern __shared__ char smem[];
    _Float16* wl   = (_Float16*)smem;                                    // [512][WSTR], K 0..255
    _Float16* hbuf = (_Float16*)(smem + RNN_H * WSTR * 2);               // [2][BQ][HSTR]
    float*    winl = (float*)(smem + RNN_H * WSTR * 2 + 2 * BQ * HSTR * 2);
    float*    xb   = winl + RNN_H;                                       // [2][BQ]

    const int tid  = threadIdx.x;
    const int lane = tid & 31;
    const int wv   = tid >> 5;              // wave 0..15
    const int b0   = blockIdx.x * BQ;

    // ---- one-time fills: W_rec K-half into LDS (16B chunks), w_in, h=0, x_0
    for (int c = tid; c < RNN_H * (KHALF / 8); c += NTHREADS) {
        int n  = c >> 5;                    // 32 chunks of 8 halves per row
        int k8 = (c & 31) << 3;
        *(v8h*)(wl + n * WSTR + k8) = *(const v8h*)(wh + n * RNN_H + k8);
    }
    for (int i = tid; i < RNN_H; i += NTHREADS) winl[i] = win[i];
    for (int i = tid; i < BQ * HSTR; i += NTHREADS) hbuf[i] = (_Float16)0.0f;
    if (tid < BQ) xb[tid] = xp[b0 + tid];
    __syncthreads();

    const int lm = lane & 15;               // M (A) / N (B) within tile
    const int ah = (lane >> 4) << 3;        // D row offset: 0 or 8
    const int bh = (lane >> 4) << 4;        // B K-phase: 0 or 16

    const _Float16* blp[2];                 // LDS B base (K-half 0)
    const _Float16* bgp[2];                 // global B base (K-half 1)
#pragma unroll
    for (int j = 0; j < 2; ++j) {
        const int n = (wv * 2 + j) * 16 + lm;
        blp[j] = wl + n * WSTR + bh;
        bgp[j] = wh + n * RNN_H + KHALF + bh;
    }

    for (int t = 0; t < RNN_T; ++t) {
        asm volatile("" ::: "memory");      // stop LICM hoisting B frags across t (spill guard)
        const int cur = t & 1, nxt = cur ^ 1;
        const _Float16* hc = hbuf + cur * BQ * HSTR;
        _Float16*       hn = hbuf + nxt * BQ * HSTR;

        v8f acc[2] = {};

        // prefetch streamed K-half B frags from L2 up front (overlaps LDS-half WMMAs)
        v16h bg[2][8];
#pragma unroll
        for (int kc = 0; kc < 8; ++kc) {
            bg[0][kc] = *(const v16h*)(bgp[0] + kc * 32);
            bg[1][kc] = *(const v16h*)(bgp[1] + kc * 32);
        }

        // K-half 0: B from LDS
#pragma unroll
        for (int kc = 0; kc < 8; ++kc) {
            const _Float16* ap = hc + lm * HSTR + kc * 32 + bh;
            HV a;
            a.v8[0] = *(const v8h*)(ap);
            a.v8[1] = *(const v8h*)(ap + 8);
#pragma unroll
            for (int j = 0; j < 2; ++j) {
                v16h b = *(const v16h*)(blp[j] + kc * 32);
                acc[j] = __builtin_amdgcn_wmma_f32_16x16x32_f16(
                    false, a.v16, false, b, (short)0, acc[j], false, false);
            }
        }
        // K-half 1: B from prefetch registers
#pragma unroll
        for (int kc = 0; kc < 8; ++kc) {
            const _Float16* ap = hc + lm * HSTR + (kc + 8) * 32 + bh;
            HV a;
            a.v8[0] = *(const v8h*)(ap);
            a.v8[1] = *(const v8h*)(ap + 8);
#pragma unroll
            for (int j = 0; j < 2; ++j)
                acc[j] = __builtin_amdgcn_wmma_f32_16x16x32_f16(
                    false, a.v16, false, bg[j][kc], (short)0, acc[j], false, false);
        }

        // epilogue: h_new = relu(acc + x_t[row]*w_in[col]) -> f16 into other buffer
        const float* xc = xb + cur * BQ;
#pragma unroll
        for (int j = 0; j < 2; ++j) {
            const int col = (wv * 2 + j) * 16 + lm;
            const float wv_in = winl[col];
#pragma unroll
            for (int i = 0; i < 8; ++i) {
                const int row = i + ah;     // C/D layout: vgpr i -> row i (+8 for lanes 16..31)
                float hv = acc[j][i] + xc[row] * wv_in;
                hv = fmaxf(hv, 0.0f);
                hn[row * HSTR + col] = (_Float16)hv;
            }
        }
        if (tid < BQ && t + 1 < RNN_T)
            xb[nxt * BQ + tid] = xp[(t + 1) * RNN_B + b0 + tid];
        __syncthreads();
    }

    // ---- output head: logits = h_last @ W_out.T + b_out (T even -> buffer 0)
    const _Float16* hf = hbuf;
    for (int idx = tid; idx < BQ * RNN_C; idx += NTHREADS) {
        const int bl = idx / RNN_C, c = idx % RNN_C;
        float s = 0.0f;
        for (int k = 0; k < RNN_H; ++k)
            s += (float)hf[bl * HSTR + k] * wout[c * RNN_H + k];
        out[(b0 + bl) * RNN_C + c] = s + bout[c];
    }
}

extern "C" void kernel_launch(void* const* d_in, const int* in_sizes, int n_in,
                              void* d_out, int out_size, void* d_ws, size_t ws_size,
                              hipStream_t stream) {
    const float* inputs = (const float*)d_in[0];
    const float* w_in   = (const float*)d_in[1];
    const float* w_rec  = (const float*)d_in[2];
    const float* w_out  = (const float*)d_in[3];
    const float* b_out  = (const float*)d_in[4];
    const int*   perm   = (const int*)d_in[5];
    float* out = (float*)d_out;

    _Float16* wh = (_Float16*)d_ws;                                   // 512 KB
    float*    xp = (float*)((char*)d_ws + (size_t)RNN_H * RNN_H * 2); // 3.14 MB

    wrec_to_f16<<<(RNN_H * RNN_H + 255) / 256, 256, 0, stream>>>(w_rec, wh);
    permute_x  <<<(RNN_T * RNN_B + 255) / 256, 256, 0, stream>>>(inputs, perm, xp);

    const size_t smem = (size_t)RNN_H * WSTR * sizeof(_Float16)   // W_rec K-half
                      + 2 * BQ * HSTR * sizeof(_Float16)          // h double buffer
                      + RNN_H * sizeof(float)                     // w_in
                      + 2 * BQ * sizeof(float);                   // x double buffer
    rnn_kernel<<<RNN_B / BQ, NTHREADS, smem, stream>>>(xp, wh, w_in, w_out, b_out, out);
}
